// CrossAttention_49692771615259
// MI455X (gfx1250) — compile-verified, matrix operand order checked by reading, not run on hardware
//
#include <hip/hip_runtime.h>
#include <hip/hip_bf16.h>

// ---------------------------------------------------------------------------
// CrossAttention on MI455X (gfx1250): all matmuls via v_wmma_f32_16x16x32_bf16
// Q-tile staged to LDS via the Tensor Data Mover (tensor_load_to_lds).
// ---------------------------------------------------------------------------

#define NB    8        // batch
#define EMB   512
#define CIN   512
#define LQ    4096     // 64*64
#define LK    1024     // 32*32

typedef __bf16        bf16_t;
typedef __bf16        v16bf __attribute__((ext_vector_type(16)));
typedef float         v8f   __attribute__((ext_vector_type(8)));
typedef unsigned int  u32x4 __attribute__((ext_vector_type(4)));
typedef int           i32x4 __attribute__((ext_vector_type(4)));
typedef int           i32x8 __attribute__((ext_vector_type(8)));

#if defined(__has_builtin)
#if __has_builtin(__builtin_amdgcn_sched_barrier)
#define SCHED_FENCE() __builtin_amdgcn_sched_barrier(0)
#endif
#endif
#ifndef SCHED_FENCE
#define SCHED_FENCE()
#endif

// ---- helpers ---------------------------------------------------------------

__device__ __forceinline__ unsigned short f32_to_bf16(float f) {
  unsigned int u = __float_as_uint(f);
  unsigned int r = (u + 0x7FFFu + ((u >> 16) & 1u)) >> 16;   // RNE
  return (unsigned short)r;
}

union Frag16 { v16bf v; u32x4 q[2]; };

// Load a 16x32 bf16 operand tile (row-major, leading dim `ld` elements) into
// the wave32 WMMA A/B register layout:
//   lanes 0-15 : row = lane,    K = 0..7 (q0), 16..23 (q1)
//   lanes16-31 : row = lane-16, K = 8..15 (q0), 24..31 (q1)
__device__ __forceinline__ v16bf load_frag_rowK(const unsigned short* base, int ld) {
  int lane = threadIdx.x & 31;
  int r    = lane & 15;
  int hi8  = (lane >> 4) << 3;            // 0 or 8
  const unsigned short* p = base + (size_t)r * ld + hi8;
  Frag16 f;
  f.q[0] = *(const u32x4*)(p);
  f.q[1] = *(const u32x4*)(p + 16);
  return f.v;
}

__device__ __forceinline__ v8f wmma_bf16(v16bf a, v16bf b, v8f c) {
  return __builtin_amdgcn_wmma_f32_16x16x32_bf16(
      /*neg_a=*/false, a, /*neg_b=*/false, b,
      /*c_mod=*/(short)0, c, /*reuse_a=*/false, /*reuse_b=*/false);
}

// ---- Tensor Data Mover: 2-D tile load, global -> LDS -----------------------
// Packs a D# (descriptor) per CDNA5 ISA §8.3/8.4 and issues TENSOR_LOAD_TO_LDS.
// data elements are 2 bytes (bf16): data_size code = 1.
__device__ __forceinline__ void tdm_load_2d_to_lds(
    unsigned int lds_byte_off, const void* gaddr,
    unsigned int tensor_d0, unsigned int tensor_d1,   // tensor extents (elements)
    unsigned int tile_d0,   unsigned int tile_d1,     // tile extents   (elements)
    unsigned int stride_d0)                           // dim0 stride    (elements)
{
  unsigned long long ga = (unsigned long long)gaddr;
  u32x4 g0;
  g0[0] = 1u;                                            // count=1, user mode
  g0[1] = lds_byte_off;                                  // lds_addr  [63:32]
  g0[2] = (unsigned int)(ga & 0xFFFFFFFFu);              // global_addr[31:0]
  g0[3] = (unsigned int)((ga >> 32) & 0x01FFFFFFu)       // global_addr[56:32]
        | (2u << 30);                                    // type=2 ("image")
  i32x8 g1;
  g1[0] = (int)(1u << 16);                               // wg_mask=0, data_size=2B
  g1[1] = (int)((tensor_d0 & 0xFFFFu) << 16);            // tensor_dim0[15:0]
  g1[2] = (int)((tensor_d0 >> 16) | ((tensor_d1 & 0xFFFFu) << 16));
  g1[3] = (int)((tensor_d1 >> 16) | (tile_d0 << 16));    // tile_dim0
  g1[4] = (int)(tile_d1);                                // tile_dim1, tile_dim2=0
  g1[5] = (int)(stride_d0);                              // tensor_dim0_stride lo
  g1[6] = 0;
  g1[7] = 0;
  i32x4 z4 = (i32x4)0;
#if __clang_major__ >= 23
  i32x8 z8 = (i32x8)0;
  __builtin_amdgcn_tensor_load_to_lds(g0, g1, z4, z4, z8, 0);
#else
  __builtin_amdgcn_tensor_load_to_lds(g0, g1, z4, z4, 0);
#endif
}

// ---- prep kernels ----------------------------------------------------------

// Weff[e][c] = sum_j Wq[e][j]*Wpi[j][c] (bf16);  beff[e] = bq[e]+sum_j Wq[e][j]*bpi[j]
__global__ __launch_bounds__(64)
void k_combine(const float* __restrict__ Wq, const float* __restrict__ Wpi,
               const float* __restrict__ bq, const float* __restrict__ bpi,
               unsigned short* __restrict__ Weff, float* __restrict__ beff) {
  int e = blockIdx.y;
  int c = blockIdx.x * 64 + threadIdx.x;
  float s = 0.f;
  for (int j = 0; j < CIN; ++j) s = fmaf(Wq[e * CIN + j], Wpi[j * CIN + c], s);
  Weff[e * CIN + c] = f32_to_bf16(s);
  if (blockIdx.x == 0 && threadIdx.x == 0) {
    float t = bq[e];
    for (int j = 0; j < CIN; ++j) t = fmaf(Wq[e * CIN + j], bpi[j], t);
    beff[e] = t;
  }
}

__global__ __launch_bounds__(256)
void k_convert(const float* __restrict__ in, unsigned short* __restrict__ out, int n) {
  for (int i = blockIdx.x * blockDim.x + threadIdx.x; i < n; i += gridDim.x * blockDim.x)
    out[i] = f32_to_bf16(in[i]);
}

// [B, C, L] f32 -> [B, L, C] bf16 (tiled transpose)
__global__ __launch_bounds__(256)
void k_transpose_to_bf16(const float* __restrict__ in, unsigned short* __restrict__ out,
                         int C, int L) {
  __shared__ float tile[32][33];
  int b  = blockIdx.z;
  int c0 = blockIdx.y * 32;
  int l0 = blockIdx.x * 32;
  int tx = threadIdx.x, ty = threadIdx.y;      // (32,8)
  for (int i = ty; i < 32; i += 8)
    tile[i][tx] = in[((size_t)b * C + c0 + i) * L + l0 + tx];
  __syncthreads();
  for (int i = ty; i < 32; i += 8)
    out[((size_t)b * L + l0 + i) * C + c0 + tx] = f32_to_bf16(tile[tx][i]);
}

// ---- generic bf16 WMMA GEMM: C = A @ Bt^T + bias ---------------------------
// A: [M,K] bf16 row-major; Bt: [N,K] bf16 row-major; bias: [N] f32.
// OUT_MODE 0: bf16 row-major [M,N]
// OUT_MODE 1: bf16 batch-transposed: m=b*Lper+l -> out[(b*N+n)*Lper+l]
// OUT_MODE 2: f32  batch-transposed (final [B,C,H,W] output)
// Register double-buffered over BK so HBM loads overlap the WMMA burst.
template <int OUT_MODE>
__global__ __launch_bounds__(256)
void k_gemm(const unsigned short* __restrict__ A, const unsigned short* __restrict__ Bt,
            const float* __restrict__ bias, void* __restrict__ Cout,
            int M, int N, int K, int Lper) {
  constexpr int BM = 128, BN = 128, BK = 32;
  __shared__ __align__(16) unsigned short As[BM * BK];
  __shared__ __align__(16) unsigned short Bs[BN * BK];

  int tid  = threadIdx.x;
  int wave = tid >> 5;
  int m0 = blockIdx.y * BM;
  int n0 = blockIdx.x * BN;

  // each thread stages two 16B chunks of A and of B per BK tile
  int r0 = tid >> 2,          c0 = (tid & 3) << 3;          // rows 0..63
  int r1 = (tid + 256) >> 2,  c1 = (tid & 3) << 3;          // rows 64..127

  const unsigned short* Ag0 = &A[(size_t)(m0 + r0) * K + c0];
  const unsigned short* Ag1 = &A[(size_t)(m0 + r1) * K + c1];
  const unsigned short* Bg0 = &Bt[(size_t)(n0 + r0) * K + c0];
  const unsigned short* Bg1 = &Bt[(size_t)(n0 + r1) * K + c1];

  u32x4 ra0 = *(const u32x4*)(Ag0);
  u32x4 ra1 = *(const u32x4*)(Ag1);
  u32x4 rb0 = *(const u32x4*)(Bg0);
  u32x4 rb1 = *(const u32x4*)(Bg1);

  v8f acc[8];
  for (int i = 0; i < 8; ++i) acc[i] = (v8f)(0.0f);

  for (int kt = 0; kt < K; kt += BK) {
    *(u32x4*)&As[r0 * BK + c0] = ra0;
    *(u32x4*)&As[r1 * BK + c1] = ra1;
    *(u32x4*)&Bs[r0 * BK + c0] = rb0;
    *(u32x4*)&Bs[r1 * BK + c1] = rb1;
    __syncthreads();

    if (kt + BK < K) {                      // issue next tile's loads now;
      ra0 = *(const u32x4*)(Ag0 + kt + BK); // they complete under the WMMAs
      ra1 = *(const u32x4*)(Ag1 + kt + BK);
      rb0 = *(const u32x4*)(Bg0 + kt + BK);
      rb1 = *(const u32x4*)(Bg1 + kt + BK);
      __builtin_prefetch(Ag0 + kt + 2 * BK, 0, 0);
      __builtin_prefetch(Bg0 + kt + 2 * BK, 0, 0);
    }

    v16bf af = load_frag_rowK(&As[(wave * 16) * BK], BK);
    v16bf bfr[8];
    for (int nf = 0; nf < 8; ++nf)
      bfr[nf] = load_frag_rowK(&Bs[(nf * 16) * BK], BK);
    SCHED_FENCE();
    for (int nf = 0; nf < 8; ++nf)
      acc[nf] = wmma_bf16(af, bfr[nf], acc[nf]);
    __syncthreads();
  }

  // epilogue: C/D layout -> lane n = lane&15, rows m = ((lane>>4)<<3)+r
  int lane = tid & 31;
  int nin  = lane & 15;
  int mb   = (lane >> 4) << 3;
  for (int nf = 0; nf < 8; ++nf) {
    int gn = n0 + nf * 16 + nin;
    float bv = bias[gn];
    for (int r = 0; r < 8; ++r) {
      int gm = m0 + wave * 16 + mb + r;
      float val = acc[nf][r] + bv;
      if (OUT_MODE == 0) {
        ((unsigned short*)Cout)[(size_t)gm * N + gn] = f32_to_bf16(val);
      } else {
        int bb = gm / Lper, l = gm - bb * Lper;
        size_t o = ((size_t)bb * N + gn) * Lper + l;
        if (OUT_MODE == 1) ((unsigned short*)Cout)[o] = f32_to_bf16(val);
        else               ((float*)Cout)[o] = val;
      }
    }
  }
}

// ---- fused attention: per block = one batch, 32 query rows, all Lk ---------
// Qb [B,LQ,D] bf16; Kb [B,LK,D] bf16; Vt [B,D,LK] bf16 (transposed);
// Ob [B,LQ,D] bf16.  LDS: Q tile 32KB + S 128KB (f32) + P 64KB (bf16) + red.
// Each wave keeps BOTH 16-row accumulator strips live so every K/V fragment
// fetched from memory feeds two WMMAs (halves B-operand traffic).
__global__ __launch_bounds__(256)
void k_attention(const unsigned short* __restrict__ Qb, const unsigned short* __restrict__ Kb,
                 const unsigned short* __restrict__ Vt, unsigned short* __restrict__ Ob,
                 float scale) {
  constexpr int D = EMB, TM = 32;
  extern __shared__ __align__(16) char smem_raw[];
  unsigned short* Qs = (unsigned short*)smem_raw;                        //  32768 B
  float*          Ss = (float*)(smem_raw + 32768);                       // 131072 B
  unsigned short* Ps = (unsigned short*)(smem_raw + 32768 + 131072);     //  65536 B
  float*         red = (float*)(smem_raw + 32768 + 131072 + 65536);      //   1024 B
  float*      rowmax = red + 256;
  float*      rowsum = rowmax + 32;

  int b  = blockIdx.y;
  int m0 = blockIdx.x * TM;
  int tid = threadIdx.x, wave = tid >> 5, lane = tid & 31;
  int nin = lane & 15, mb = (lane >> 4) << 3;

  // ---- Q tile (32 x 512 bf16 = 32KB) via Tensor Data Mover -----------------
  const unsigned short* Qg = Qb + ((size_t)b * LQ + m0) * D;
  if (wave == 0) {
    // dynamic LDS region starts at offset 0 of this workgroup's allocation
    tdm_load_2d_to_lds(/*lds_off=*/0u, Qg,
                       /*tensor_d0=*/D, /*tensor_d1=*/(unsigned)(NB * LQ - m0),
                       /*tile_d0=*/D, /*tile_d1=*/TM,
                       /*stride_d0=*/D);
    __builtin_amdgcn_s_wait_tensorcnt(0);
  }
  __syncthreads();

  // ---- S = scale * Q K^T ; wave w owns columns [w*128, w*128+128) ----------
  // Both 16-row strips accumulated in-wave: each K fragment is used twice.
  const unsigned short* Kg = Kb + (size_t)b * LK * D;
  {
    v8f acc0[8], acc1[8];
    for (int i = 0; i < 8; ++i) { acc0[i] = (v8f)(0.0f); acc1[i] = (v8f)(0.0f); }
    for (int kk = 0; kk < D / 32; ++kk) {
      v16bf af0 = load_frag_rowK(Qs + kk * 32, D);            // rows 0..15
      v16bf af1 = load_frag_rowK(Qs + 16 * D + kk * 32, D);   // rows 16..31
      for (int g = 0; g < 2; ++g) {                           // 2 groups of 4 cols
        v16bf bfr[4];
        for (int j = 0; j < 4; ++j) {
          int ncol = wave * 128 + g * 64 + j * 16;
          bfr[j] = load_frag_rowK(Kg + (size_t)ncol * D + kk * 32, D);
        }
        SCHED_FENCE();
        for (int j = 0; j < 4; ++j) {
          acc0[g * 4 + j] = wmma_bf16(af0, bfr[j], acc0[g * 4 + j]);
          acc1[g * 4 + j] = wmma_bf16(af1, bfr[j], acc1[g * 4 + j]);
        }
      }
    }
    for (int cs = 0; cs < 8; ++cs) {
      int ncol = wave * 128 + cs * 16 + nin;
      for (int r = 0; r < 8; ++r) {
        Ss[(size_t)(mb + r) * LK + ncol]      = acc0[cs][r] * scale;
        Ss[(size_t)(16 + mb + r) * LK + ncol] = acc1[cs][r] * scale;
      }
    }
  }
  __syncthreads();

  // ---- row softmax: 8 threads per row, 128 cols each -----------------------
  {
    int row = tid >> 3, seg = tid & 7;
    const float* srow = Ss + (size_t)row * LK + seg * 128;
    float mx = -3.0e38f;
    for (int i = 0; i < 128; ++i) mx = fmaxf(mx, srow[i]);
    red[row * 8 + seg] = mx;
    __syncthreads();
    if (seg == 0) {
      float m2 = red[row * 8];
      for (int i = 1; i < 8; ++i) m2 = fmaxf(m2, red[row * 8 + i]);
      rowmax[row] = m2;
    }
    __syncthreads();
    float rm = rowmax[row], sm = 0.f;
    for (int i = 0; i < 128; ++i) sm += __expf(srow[i] - rm);
    red[row * 8 + seg] = sm;
    __syncthreads();
    if (seg == 0) {
      float s2 = 0.f;
      for (int i = 0; i < 8; ++i) s2 += red[row * 8 + i];
      rowsum[row] = 1.0f / s2;
    }
    __syncthreads();
  }
  // P = exp(S - max) / sum  -> bf16
  for (int i = tid; i < TM * LK; i += 256) {
    int row = i >> 10;  // / LK
    Ps[i] = f32_to_bf16(__expf(Ss[i] - rowmax[row]) * rowsum[row]);
  }
  __syncthreads();

  // ---- O = P @ V ; wave w owns a 32 x 64 output block ----------------------
  // Both row strips in-wave: each V fragment is used twice.
  const unsigned short* Vg = Vt + (size_t)b * D * LK;
  v8f acc0[4], acc1[4];
  for (int i = 0; i < 4; ++i) { acc0[i] = (v8f)(0.0f); acc1[i] = (v8f)(0.0f); }
  for (int kk = 0; kk < LK / 32; ++kk) {
    v16bf af0 = load_frag_rowK(Ps + kk * 32, LK);             // rows 0..15
    v16bf af1 = load_frag_rowK(Ps + 16 * LK + kk * 32, LK);   // rows 16..31
    v16bf bfr[4];
    for (int nf = 0; nf < 4; ++nf) {
      int n = wave * 64 + nf * 16;
      bfr[nf] = load_frag_rowK(Vg + (size_t)n * LK + kk * 32, LK);
    }
    SCHED_FENCE();
    for (int nf = 0; nf < 4; ++nf) {
      acc0[nf] = wmma_bf16(af0, bfr[nf], acc0[nf]);
      acc1[nf] = wmma_bf16(af1, bfr[nf], acc1[nf]);
    }
  }
  unsigned short* Og = Ob + ((size_t)b * LQ + m0) * D;
  for (int nf = 0; nf < 4; ++nf) {
    int gn = wave * 64 + nf * 16 + nin;
    for (int r = 0; r < 8; ++r) {
      Og[(size_t)(mb + r) * D + gn]      = f32_to_bf16(acc0[nf][r]);
      Og[(size_t)(16 + mb + r) * D + gn] = f32_to_bf16(acc1[nf][r]);
    }
  }
}

// ---------------------------------------------------------------------------

extern "C" void kernel_launch(void* const* d_in, const int* in_sizes, int n_in,
                              void* d_out, int out_size, void* d_ws, size_t ws_size,
                              hipStream_t stream) {
  const float* x    = (const float*)d_in[0];
  const float* ctx  = (const float*)d_in[1];
  const float* W_pi = (const float*)d_in[2];
  const float* b_pi = (const float*)d_in[3];
  const float* W_q  = (const float*)d_in[4];
  const float* b_q  = (const float*)d_in[5];
  const float* W_k  = (const float*)d_in[6];
  const float* b_k  = (const float*)d_in[7];
  const float* W_v  = (const float*)d_in[8];
  const float* b_v  = (const float*)d_in[9];
  const float* W_po = (const float*)d_in[10];
  const float* b_po = (const float*)d_in[11];
  float* out = (float*)d_out;

  // workspace carve-up (all 256B aligned); total ~128 MB
  char* ws = (char*)d_ws;
  size_t off = 0;
  auto carve = [&](size_t bytes) { char* p = ws + off; off = (off + bytes + 255) & ~(size_t)255; return p; };
  unsigned short* Xb   = (unsigned short*)carve((size_t)NB * LQ * CIN * 2);  // [B,LQ,CIN]
  unsigned short* Ctx  = (unsigned short*)carve((size_t)NB * LK * EMB * 2);  // [B,LK,EMB]
  unsigned short* Weff = (unsigned short*)carve((size_t)EMB * CIN * 2);
  unsigned short* Wkb  = (unsigned short*)carve((size_t)EMB * EMB * 2);
  unsigned short* Wvb  = (unsigned short*)carve((size_t)EMB * EMB * 2);
  unsigned short* Wpob = (unsigned short*)carve((size_t)CIN * EMB * 2);
  float*          beff = (float*)carve((size_t)EMB * 4);
  unsigned short* Qb   = (unsigned short*)carve((size_t)NB * LQ * EMB * 2);
  unsigned short* Kb   = (unsigned short*)carve((size_t)NB * LK * EMB * 2);
  unsigned short* Vtb  = (unsigned short*)carve((size_t)NB * EMB * LK * 2);  // transposed
  unsigned short* Ob   = (unsigned short*)carve((size_t)NB * LQ * EMB * 2);

  // 1) weight prep
  k_combine<<<dim3(CIN / 64, EMB), 64, 0, stream>>>(W_q, W_pi, b_q, b_pi, Weff, beff);
  k_convert<<<256, 256, 0, stream>>>(W_k, Wkb, EMB * EMB);
  k_convert<<<256, 256, 0, stream>>>(W_v, Wvb, EMB * EMB);
  k_convert<<<256, 256, 0, stream>>>(W_po, Wpob, CIN * EMB);

  // 2) activation layout: [B,C,L] -> [B,L,C] bf16
  k_transpose_to_bf16<<<dim3(LQ / 32, CIN / 32, NB), dim3(32, 8), 0, stream>>>(x, Xb, CIN, LQ);
  k_transpose_to_bf16<<<dim3(LK / 32, EMB / 32, NB), dim3(32, 8), 0, stream>>>(ctx, Ctx, EMB, LK);

  // 3) projections (WMMA GEMMs); M spans batches since weights are shared
  k_gemm<0><<<dim3(EMB / 128, (NB * LQ) / 128), 256, 0, stream>>>(
      Xb, Weff, beff, Qb, NB * LQ, EMB, CIN, 0);
  k_gemm<0><<<dim3(EMB / 128, (NB * LK) / 128), 256, 0, stream>>>(
      Ctx, Wkb, b_k, Kb, NB * LK, EMB, EMB, 0);
  k_gemm<1><<<dim3(EMB / 128, (NB * LK) / 128), 256, 0, stream>>>(
      Ctx, Wvb, b_v, Vtb, NB * LK, EMB, EMB, LK);  // store V transposed

  // 4) fused attention (needs 320KB-LDS-class WGP: ~226KB dynamic LDS)
  const int SMEM = 32768 + 131072 + 65536 + 2048;
  hipFuncSetAttribute((const void*)k_attention,
                      hipFuncAttributeMaxDynamicSharedMemorySize, SMEM);
  const float scale = 0.044194173824159216f;  // 512^-0.5
  k_attention<<<dim3(LQ / 32, NB), 256, SMEM, stream>>>(Qb, Kb, Vtb, Ob, scale);

  // 5) proj_out, writing f32 directly in [B,C,H,W]
  k_gemm<2><<<dim3(CIN / 128, (NB * LQ) / 128), 256, 0, stream>>>(
      Ob, Wpob, b_po, out, NB * LQ, CIN, EMB, LQ);
}
